// SmileMoELinear_87436944212180
// MI455X (gfx1250) — matrix operand order repaired and checked by reading, not compiled, not run on hardware
//
#include <hip/hip_runtime.h>
#include <hip/hip_bf16.h>

#define T_TOK   4096
#define NEXP    8
#define DIN     2048
#define DOUT    2048
#define BM      128         // tokens per block tile (8 waves x 16)
#define BN      64          // output cols per block tile
#define BK      32          // K per WMMA step
#define LDST    48          // padded LDS row stride in bf16 units (32 data + 16 pad), keeps 16B alignment

typedef __attribute__((ext_vector_type(16))) __bf16 v16bf;
typedef __attribute__((ext_vector_type(8)))  float  v8f;

// pack hi16(a) into low half, hi16(b) into high half: one v_perm_b32
__device__ __forceinline__ unsigned int pack_hi16(float a, float b) {
    return __builtin_amdgcn_perm(__float_as_uint(b), __float_as_uint(a), 0x07060302u);
}

// truncation-based bf16 hi/lo split of 4 floats; lo is the bf16-truncated residual v - hi
__device__ __forceinline__ void cvt_store4(unsigned short* hdst, unsigned short* ldst,
                                           float v0, float v1, float v2, float v3) {
    unsigned int u0 = __float_as_uint(v0), u1 = __float_as_uint(v1);
    unsigned int u2 = __float_as_uint(v2), u3 = __float_as_uint(v3);
    *reinterpret_cast<unsigned int*>(hdst)     = pack_hi16(v0, v1);
    *reinterpret_cast<unsigned int*>(hdst + 2) = pack_hi16(v2, v3);
    float r0 = v0 - __uint_as_float(u0 & 0xffff0000u);
    float r1 = v1 - __uint_as_float(u1 & 0xffff0000u);
    float r2 = v2 - __uint_as_float(u2 & 0xffff0000u);
    float r3 = v3 - __uint_as_float(u3 & 0xffff0000u);
    *reinterpret_cast<unsigned int*>(ldst)     = pack_hi16(r0, r1);
    *reinterpret_cast<unsigned int*>(ldst + 2) = pack_hi16(r2, r3);
}

// ---------------- kernel 0: zero output + expert counters ----------------
__global__ void moe_zero_kernel(float* __restrict__ out, int n, int* __restrict__ cnt) {
    int i = blockIdx.x * blockDim.x + threadIdx.x;
    if (i < NEXP) cnt[i] = 0;
    for (int j = i; j < n; j += gridDim.x * blockDim.x) out[j] = 0.0f;
}

// ---------------- kernel 1: softmax + top-2 gating, scatter to expert lists ----------------
__global__ void moe_gate_kernel(const float* __restrict__ logits,
                                int* __restrict__ cnt,
                                int* __restrict__ idx,
                                float* __restrict__ wl) {
    int t = blockIdx.x * blockDim.x + threadIdx.x;
    if (t >= T_TOK) return;
    float l[NEXP];
    float m = -3.402823e38f;
    #pragma unroll
    for (int e = 0; e < NEXP; e++) { l[e] = logits[t * NEXP + e]; m = fmaxf(m, l[e]); }
    float p[NEXP];
    #pragma unroll
    for (int e = 0; e < NEXP; e++) p[e] = expf(l[e] - m);
    // top-2 (earlier index wins ties, matching lax.top_k)
    int i0 = 0;
    #pragma unroll
    for (int e = 1; e < NEXP; e++) if (p[e] > p[i0]) i0 = e;
    int i1 = (i0 == 0) ? 1 : 0;
    #pragma unroll
    for (int e = 0; e < NEXP; e++) if (e != i0 && p[e] > p[i1]) i1 = e;
    float w0 = p[i0], w1 = p[i1];
    float inv = 1.0f / (w0 + w1);     // softmax denom cancels in the renorm
    w0 *= inv; w1 *= inv;
    int pos0 = atomicAdd(&cnt[i0], 1);
    idx[i0 * T_TOK + pos0] = t; wl[i0 * T_TOK + pos0] = w0;
    int pos1 = atomicAdd(&cnt[i1], 1);
    idx[i1 * T_TOK + pos1] = t; wl[i1 * T_TOK + pos1] = w1;
}

// ---------------- kernel 2: gathered split-bf16 WMMA GEMM ----------------
__global__ __launch_bounds__(256)
void moe_gemm_kernel(const float* __restrict__ x,
                     const float* __restrict__ W,
                     const float* __restrict__ bias,
                     const int*   __restrict__ cnt,
                     const int*   __restrict__ idx,
                     const float* __restrict__ wl,
                     float*       __restrict__ out) {
    const int e  = blockIdx.z;
    const int rt = blockIdx.x;          // token row tile within expert list
    const int nb = blockIdx.y;          // output column tile
    const int cntE = cnt[e];
    if (rt * BM >= cntE) return;

    __shared__ int   stok[BM];
    __shared__ float sw[BM];
    __shared__ __align__(16) unsigned short sxh[BM * LDST];
    __shared__ __align__(16) unsigned short sxl[BM * LDST];
    __shared__ __align__(16) unsigned short swh[BN * LDST];
    __shared__ __align__(16) unsigned short swl[BN * LDST];

    const int tid  = threadIdx.x;
    const int wv   = tid >> 5;          // 8 waves
    const int lane = tid & 31;
    const int h    = lane >> 4;
    const int ln16 = lane & 15;

    if (tid < BM) {
        int pos = rt * BM + tid;
        if (pos < cntE) { stok[tid] = idx[e * T_TOK + pos]; sw[tid] = wl[e * T_TOK + pos]; }
        else            { stok[tid] = 0;                    sw[tid] = 0.0f; }
    }
    __syncthreads();

    v8f acc[4] = {};
    const float* Wbase = W + ((size_t)e * DOUT + (size_t)nb * BN) * DIN;

    // hoist loop-invariant gather bases + LDS store offsets into registers
    const float* xptr[4];
    int xoff[4];
    #pragma unroll
    for (int q = 0; q < 4; q++) {
        int id  = q * 256 + tid;
        int row = id >> 3;
        int c4  = (id & 7) * 4;
        xptr[q] = x + (size_t)stok[row] * DIN + c4;
        xoff[q] = row * LDST + c4;
    }
    const float* wptr[2];
    int woff[2];
    #pragma unroll
    for (int q = 0; q < 2; q++) {
        int id  = q * 256 + tid;
        int row = id >> 3;
        int c4  = (id & 7) * 4;
        wptr[q] = Wbase + (size_t)row * DIN + c4;
        woff[q] = row * LDST + c4;
    }

    union Frag { uint4 u[2]; v16bf v; };
    const uint4* pxh = reinterpret_cast<const uint4*>(sxh);
    const uint4* pxl = reinterpret_cast<const uint4*>(sxl);
    const uint4* pwh = reinterpret_cast<const uint4*>(swh);
    const uint4* pwl = reinterpret_cast<const uint4*>(swl);
    const int mrowA = wv * 16 + ln16;   // A row this lane feeds

    for (int k0 = 0; k0 < DIN; k0 += BK) {
        // cooperative coalesced global loads:
        // X tile: 128 rows x 32 K = 1024 float4 -> 4 per thread
        // W tile:  64 rows x 32 K =  512 float4 -> 2 per thread
        float4 xq[4], wq[2];
        #pragma unroll
        for (int q = 0; q < 4; q++) {
            xq[q] = *reinterpret_cast<const float4*>(xptr[q] + k0);
            __builtin_prefetch(xptr[q] + k0 + BK, 0, 3);  // speculative: dropped on fault
        }
        #pragma unroll
        for (int q = 0; q < 2; q++) {
            wq[q] = *reinterpret_cast<const float4*>(wptr[q] + k0);
            __builtin_prefetch(wptr[q] + k0 + BK, 0, 3);
        }
        __syncthreads();   // previous iteration's fragment reads complete
        #pragma unroll
        for (int q = 0; q < 4; q++)
            cvt_store4(&sxh[xoff[q]], &sxl[xoff[q]], xq[q].x, xq[q].y, xq[q].z, xq[q].w);
        #pragma unroll
        for (int q = 0; q < 2; q++)
            cvt_store4(&swh[woff[q]], &swl[woff[q]], wq[q].x, wq[q].y, wq[q].z, wq[q].w);
        __syncthreads();

        // A fragment (16-bit A 16x32 layout): V0-3 = K 8h+0..7, V4-7 = K 16+8h+0..7
        Frag axh, axl;
        axh.u[0] = pxh[mrowA * 6 + h];     axh.u[1] = pxh[mrowA * 6 + 2 + h];
        axl.u[0] = pxl[mrowA * 6 + h];     axl.u[1] = pxl[mrowA * 6 + 2 + h];

        #pragma unroll
        for (int ns = 0; ns < 4; ns++) {
            // B fragment (16-bit B 32x16 layout): V0-3 = K 8h+0..7, V4-7 = K 16+8h+0..7
            int n = ns * 16 + ln16;
            Frag bh, bl;
            bh.u[0] = pwh[n * 6 + h];  bh.u[1] = pwh[n * 6 + 2 + h];
            bl.u[0] = pwl[n * 6 + h];  bl.u[1] = pwl[n * 6 + 2 + h];
            // split-bf16: x.w ~= xh.wh + xh.wl + xl.wh  (fp32 accumulate)
            acc[ns] = __builtin_amdgcn_wmma_f32_16x16x32_bf16(false, axh.v, false, bh.v, (short)0, acc[ns], false, false);
            acc[ns] = __builtin_amdgcn_wmma_f32_16x16x32_bf16(false, axh.v, false, bl.v, (short)0, acc[ns], false, false);
            acc[ns] = __builtin_amdgcn_wmma_f32_16x16x32_bf16(false, axl.v, false, bh.v, (short)0, acc[ns], false, false);
        }
    }

    // epilogue: gate weight * (dot + bias), commutative f32 atomics -> deterministic
    const float* bptr = bias + (size_t)e * DOUT + (size_t)nb * BN;
    const int validRows = cntE - rt * BM;
    #pragma unroll
    for (int ns = 0; ns < 4; ns++) {
        #pragma unroll
        for (int r = 0; r < 8; r++) {
            int mrow = wv * 16 + r + h * 8;      // C/D layout: lanes16-31 hold M+8
            if (mrow < validRows) {
                int n = ns * 16 + ln16;
                float v = sw[mrow] * (acc[ns][r] + bptr[n]);
                atomicAdd(out + (size_t)stok[mrow] * DOUT + (size_t)nb * BN + n, v);
            }
        }
    }
}

extern "C" void kernel_launch(void* const* d_in, const int* in_sizes, int n_in,
                              void* d_out, int out_size, void* d_ws, size_t ws_size,
                              hipStream_t stream) {
    const float* x      = (const float*)d_in[0];   // [B,S,DIN] fp32
    const float* logits = (const float*)d_in[1];   // [T,E] fp32
    const float* W      = (const float*)d_in[2];   // [E,DOUT,DIN] fp32
    const float* b      = (const float*)d_in[3];   // [E,DOUT] fp32
    float* out = (float*)d_out;

    // workspace layout
    int*   cnt = (int*)d_ws;                                           // 8 ints
    int*   idx = (int*)((char*)d_ws + 64);                             // E*T ints
    float* wl  = (float*)((char*)d_ws + 64 + (size_t)NEXP * T_TOK * sizeof(int));

    moe_zero_kernel<<<1024, 256, 0, stream>>>(out, T_TOK * DOUT, cnt);
    moe_gate_kernel<<<T_TOK / 256, 256, 0, stream>>>(logits, cnt, idx, wl);
    dim3 grid(T_TOK / BM, DOUT / BN, NEXP);   // row tiles over-provisioned; blocks exit on cnt[e]
    moe_gemm_kernel<<<grid, 256, 0, stream>>>(x, W, b, cnt, idx, wl, out);
}
